// VisionMultiHeadAttention_50663434223684
// MI455X (gfx1250) — matrix operand order repaired and checked by reading, not compiled, not run on hardware
//
#include <hip/hip_runtime.h>

typedef _Float16 half_t;
typedef __attribute__((ext_vector_type(16))) _Float16 v16h;
typedef __attribute__((ext_vector_type(8)))  _Float16 v8h;
typedef __attribute__((ext_vector_type(8)))  float    v8f;

// ---------------------------------------------------------------------------
// WMMA fragment helpers (layouts per CDNA5 ISA 7.12.2, wave32)
// A (16x32 f16): lane l (m=l&15, g=l>>4) holds row m, K chunks [k0+8g,+8) and
//                [k0+16+8g,+8)  -> two 16B contiguous loads from row-major A.
// B (32x16 f16): lane l (n=l&15, g=l>>4) holds col n, K chunks [k0+8g,+8) and
//                [k0+16+8g,+8) -> two 16B loads from N-major (K-contiguous) Bt.
// C/D (16x16 f32): lane l col n=l&15, VGPR r row m=8*(l>>4)+r.
// ---------------------------------------------------------------------------

static __device__ inline v8f wmma_f16(v16h a, v16h b, v8f c) {
  return __builtin_amdgcn_wmma_f32_16x16x32_f16(false, a, false, b, (short)0, c,
                                                false, false);
}

static __device__ inline v16h load_a_frag(const half_t* __restrict__ A, int lda,
                                          int row0, int k0) {
  const int l = threadIdx.x & 31;
  const int m = l & 15, g = l >> 4;
  const half_t* p = A + (size_t)(row0 + m) * lda + k0 + 8 * g;
  v8h lo = *reinterpret_cast<const v8h*>(p);
  v8h hi = *reinterpret_cast<const v8h*>(p + 16);
  v16h a;
#pragma unroll
  for (int i = 0; i < 8; ++i) { a[i] = lo[i]; a[i + 8] = hi[i]; }
  return a;
}

static __device__ inline v16h load_b_frag(const half_t* __restrict__ Bt, int ldb,
                                          int n0, int k0) {
  const int l = threadIdx.x & 31;
  const int n = l & 15, g = l >> 4;
  const half_t* p = Bt + (size_t)(n0 + n) * ldb + k0 + 8 * g;
  v8h lo = *reinterpret_cast<const v8h*>(p);
  v8h hi = *reinterpret_cast<const v8h*>(p + 16);
  v16h b;
#pragma unroll
  for (int i = 0; i < 8; ++i) { b[i] = lo[i]; b[i + 8] = hi[i]; }
  return b;
}

// ---------------------------------------------------------------------------
// Conversion / transpose kernels
// ---------------------------------------------------------------------------
__global__ void cvt_f32_to_f16(const float* __restrict__ in,
                               half_t* __restrict__ out, int n) {
  int i = blockIdx.x * blockDim.x + threadIdx.x;
  if (i < n) out[i] = (half_t)in[i];
}

// W [H,C,D] f32  ->  Wt [H*D, C] f16  (row n=h*D+d holds W[h,:,d])
__global__ void transpose_w_qkv(const float* __restrict__ W,
                                half_t* __restrict__ Wt, int H, int C, int D) {
  int i = blockIdx.x * blockDim.x + threadIdx.x;
  int total = H * D * C;
  if (i >= total) return;
  int c = i % C;
  int n = i / C;
  int h = n / D, d = n % D;
  Wt[i] = (half_t)W[((size_t)h * C + c) * D + d];
}

// W [C,C] f32 -> Wt [N=C][K=C] f16 with Wt[n][c] = W[c][n]
__global__ void transpose_proj(const float* __restrict__ W,
                               half_t* __restrict__ Wt, int C) {
  int i = blockIdx.x * blockDim.x + threadIdx.x;
  if (i >= C * C) return;
  int c = i % C;
  int n = i / C;
  Wt[i] = (half_t)W[(size_t)c * C + n];
}

// ---------------------------------------------------------------------------
// Generic WMMA GEMM: per wave, 16(M) x 64(N) tile, K-loop step 32.
//   mode 0: store f16 in head layout  [B,H,T,D]   (for Q and K)
//   mode 1: store f16 V-transposed    [B,H,D,T]
//   mode 2: store f32 row-major [M,N] + bias      (final projection)
// ---------------------------------------------------------------------------
__global__ __launch_bounds__(256) void gemm_wmma(
    const half_t* __restrict__ A, const half_t* __restrict__ Bt,
    void* __restrict__ outp, const float* __restrict__ bias, int M, int N,
    int K, int mode) {
  const int wave = blockIdx.x * (blockDim.x >> 5) + (threadIdx.x >> 5);
  const int tilesN = N >> 6;
  const int mi = wave / tilesN, ni = wave % tilesN;
  const int row0 = mi << 4, col0 = ni << 6;
  if (row0 >= M) return;  // uniform per wave -> EXEC stays all-ones

  const int l = threadIdx.x & 31;
  const int g = l >> 4, lane_n = l & 15;

  v8f acc[4] = {};
  for (int k0 = 0; k0 < K; k0 += 32) {
    v16h a = load_a_frag(A, K, row0, k0);
#pragma unroll
    for (int j = 0; j < 4; ++j) {
      v16h b = load_b_frag(Bt, K, col0 + 16 * j, k0);
      acc[j] = wmma_f16(a, b, acc[j]);
    }
  }

  if (mode == 2) {
    float* out = (float*)outp;
#pragma unroll
    for (int j = 0; j < 4; ++j) {
      int n = col0 + 16 * j + lane_n;
      float bv = bias ? bias[n] : 0.0f;
#pragma unroll
      for (int r = 0; r < 8; ++r) {
        int bt = row0 + 8 * g + r;
        out[(size_t)bt * N + n] = acc[j][r] + bv;
      }
    }
  } else {
    half_t* out = (half_t*)outp;
#pragma unroll
    for (int j = 0; j < 4; ++j) {
      int n = col0 + 16 * j + lane_n;
      int h = n >> 6, d = n & 63;
#pragma unroll
      for (int r = 0; r < 8; ++r) {
        int bt = row0 + 8 * g + r;
        int b = bt >> 10, t = bt & 1023;
        size_t dst;
        if (mode == 0)
          dst = (((size_t)(b * 12 + h) << 10) + t) * 64 + d;  // [B,H,T,D]
        else
          dst = (((size_t)(b * 12 + h) * 64 + d) << 10) + t;  // [B,H,D,T]
        out[dst] = (half_t)acc[j][r];
      }
    }
  }
}

// ---------------------------------------------------------------------------
// Flash attention: one wave = one (b,h) and 16 query rows.
// Keys in blocks of 32 (= WMMA K). Online softmax; P restaged via LDS into
// A-fragment layout for the P@V WMMAs. Output written f16 as [B,T,H*D].
// ---------------------------------------------------------------------------
__global__ __launch_bounds__(256) void attn_wmma(
    const half_t* __restrict__ qh, const half_t* __restrict__ kh,
    const half_t* __restrict__ vth, half_t* __restrict__ attn) {
  __shared__ half_t p_lds[8][16][40];  // per-wave slice, padded rows

  const int wslot = threadIdx.x >> 5;
  const int wave = blockIdx.x * 8 + wslot;
  const int bh = wave >> 6;   // (b*H + h); 64 query tiles each
  const int qt = wave & 63;
  const int b = bh / 12, h = bh % 12;
  const int t0 = qt << 4;

  const half_t* qbase = qh + (size_t)bh * 1024 * 64;   // [T,D]
  const half_t* kbase = kh + (size_t)bh * 1024 * 64;   // [S,D] (Bt for QK^T)
  const half_t* vbase = vth + (size_t)bh * 64 * 1024;  // [D,S] (Bt for P@V)

  const int l = threadIdx.x & 31;
  const int g = l >> 4, lane_n = l & 15;

  const v16h qa0 = load_a_frag(qbase, 64, t0, 0);
  const v16h qa1 = load_a_frag(qbase, 64, t0, 32);

  float mrow[8], lrow[8];
#pragma unroll
  for (int r = 0; r < 8; ++r) { mrow[r] = -1e30f; lrow[r] = 0.0f; }
  v8f oacc[4] = {};

  half_t(*pt)[40] = p_lds[wslot];

  for (int s0 = 0; s0 < 1024; s0 += 32) {
    // S tile [16 x 32]: two 16-key column tiles, contraction over D=64
    v8f sa = {}, sb = {};
    {
      v16h bk;
      bk = load_b_frag(kbase, 64, s0, 0);       sa = wmma_f16(qa0, bk, sa);
      bk = load_b_frag(kbase, 64, s0, 32);      sa = wmma_f16(qa1, bk, sa);
      bk = load_b_frag(kbase, 64, s0 + 16, 0);  sb = wmma_f16(qa0, bk, sb);
      bk = load_b_frag(kbase, 64, s0 + 16, 32); sb = wmma_f16(qa1, bk, sb);
    }

    // online softmax per output row r (row lives in one 16-lane half)
#pragma unroll
    for (int r = 0; r < 8; ++r) {
      float x0 = sa[r] * 0.125f;  // 1/sqrt(64)
      float x1 = sb[r] * 0.125f;
      float mx = fmaxf(x0, x1);
#pragma unroll
      for (int o = 1; o < 16; o <<= 1) mx = fmaxf(mx, __shfl_xor(mx, o, 32));
      float mnew = fmaxf(mrow[r], mx);
      float corr = __expf(mrow[r] - mnew);
      float p0 = __expf(x0 - mnew);
      float p1 = __expf(x1 - mnew);
      float ps = p0 + p1;
#pragma unroll
      for (int o = 1; o < 16; o <<= 1) ps += __shfl_xor(ps, o, 32);
      lrow[r] = lrow[r] * corr + ps;
      mrow[r] = mnew;
#pragma unroll
      for (int nd = 0; nd < 4; ++nd) oacc[nd][r] *= corr;
      // stage P into LDS in C/D layout: row 8g+r, cols lane_n / 16+lane_n
      pt[8 * g + r][lane_n] = (half_t)p0;
      pt[8 * g + r][16 + lane_n] = (half_t)p1;
    }

    // reload P as an A fragment (K = 32 keys); same-wave DS is in-order
    v16h pa;
    {
      v8h lo = *reinterpret_cast<const v8h*>(&pt[lane_n][8 * g]);
      v8h hi = *reinterpret_cast<const v8h*>(&pt[lane_n][16 + 8 * g]);
#pragma unroll
      for (int i = 0; i < 8; ++i) { pa[i] = lo[i]; pa[i + 8] = hi[i]; }
    }

    // accumulate P @ V over this key block (B operand: vt[D][S])
#pragma unroll
    for (int nd = 0; nd < 4; ++nd) {
      v16h bv = load_b_frag(vbase, 1024, nd * 16, s0);
      oacc[nd] = wmma_f16(pa, bv, oacc[nd]);
    }
  }

  // epilogue: normalize and store f16 as [B, T, H*D] (heads concatenated)
#pragma unroll
  for (int r = 0; r < 8; ++r) {
    float inv = 1.0f / lrow[r];
    int t = t0 + 8 * g + r;
    size_t rowbase = ((size_t)b * 1024 + t) * 768 + h * 64;
#pragma unroll
    for (int nd = 0; nd < 4; ++nd)
      attn[rowbase + nd * 16 + lane_n] = (half_t)(oacc[nd][r] * inv);
  }
}

// ---------------------------------------------------------------------------
extern "C" void kernel_launch(void* const* d_in, const int* in_sizes, int n_in,
                              void* d_out, int out_size, void* d_ws,
                              size_t ws_size, hipStream_t stream) {
  const float* x  = (const float*)d_in[0];
  const float* Wq = (const float*)d_in[1];
  const float* Wk = (const float*)d_in[2];
  const float* Wv = (const float*)d_in[3];
  const float* Wp = (const float*)d_in[4];
  const float* bp = (const float*)d_in[5];
  float* out = (float*)d_out;

  const int B = 8, T = 1024, C = 768, H = 12, D = 64;
  const size_t MT = (size_t)B * T;  // 8192 token rows

  char* ws = (char*)d_ws;
  size_t off = 0;
  auto alloc = [&](size_t bytes) -> void* {
    void* p = ws + off;
    off = (off + bytes + 255) & ~(size_t)255;
    return p;
  };
  half_t* xh   = (half_t*)alloc(MT * C * sizeof(half_t));
  half_t* wqt  = (half_t*)alloc((size_t)C * C * sizeof(half_t));
  half_t* wkt  = (half_t*)alloc((size_t)C * C * sizeof(half_t));
  half_t* wvt  = (half_t*)alloc((size_t)C * C * sizeof(half_t));
  half_t* wpt  = (half_t*)alloc((size_t)C * C * sizeof(half_t));
  half_t* qh   = (half_t*)alloc(MT * C * sizeof(half_t));
  half_t* kh   = (half_t*)alloc(MT * C * sizeof(half_t));
  half_t* vth  = (half_t*)alloc(MT * C * sizeof(half_t));
  half_t* atth = xh;  // alias: xh is dead once V projection completes

  // 1) f16 conversion + weight transposes
  {
    int n = (int)(MT * C);
    cvt_f32_to_f16<<<(n + 255) / 256, 256, 0, stream>>>(x, xh, n);
    int nw = H * C * D;
    transpose_w_qkv<<<(nw + 255) / 256, 256, 0, stream>>>(Wq, wqt, H, C, D);
    transpose_w_qkv<<<(nw + 255) / 256, 256, 0, stream>>>(Wk, wkt, H, C, D);
    transpose_w_qkv<<<(nw + 255) / 256, 256, 0, stream>>>(Wv, wvt, H, C, D);
    int np = C * C;
    transpose_proj<<<(np + 255) / 256, 256, 0, stream>>>(Wp, wpt, C);
  }

  // 2) QKV projections: [8192,768] x [768,768]^T -> head layouts
  // waves = (8192/16)*(768/64) = 6144 -> 768 blocks of 8 waves
  gemm_wmma<<<768, 256, 0, stream>>>(xh, wqt, qh, nullptr, (int)MT, C, C, 0);
  gemm_wmma<<<768, 256, 0, stream>>>(xh, wkt, kh, nullptr, (int)MT, C, C, 0);
  gemm_wmma<<<768, 256, 0, stream>>>(xh, wvt, vth, nullptr, (int)MT, C, C, 1);

  // 3) flash attention: waves = B*H*(T/16) = 6144 -> 768 blocks of 8 waves
  attn_wmma<<<768, 256, 0, stream>>>(qh, kh, vth, atth);

  // 4) output projection with bias -> f32 [B,T,C]
  gemm_wmma<<<768, 256, 0, stream>>>(atth, wpt, out, bp, (int)MT, C, C, 2);
}
